// ColumnConsistencyLoss_75161927680447
// MI455X (gfx1250) — matrix operand order repaired and checked by reading, not compiled, not run on hardware
//
#include <hip/hip_runtime.h>
#include <stdint.h>

typedef _Float16 v16h __attribute__((ext_vector_type(16)));
typedef _Float16 v8h  __attribute__((ext_vector_type(8)));
typedef float    v8f  __attribute__((ext_vector_type(8)));
typedef float    v4f  __attribute__((ext_vector_type(4)));
typedef unsigned int v4u __attribute__((ext_vector_type(4)));
typedef int      v8i  __attribute__((ext_vector_type(8)));
typedef int      v4i  __attribute__((ext_vector_type(4)));

#define NCOL     128   // C (columns / logit width)
#define TOK_BLK  32    // tokens per LDS tile (== WMMA K)
#define PT_STR   40    // probsT row stride in halves (32 + 8 pad -> conflict-free b128 reads)

// ---------------------------------------------------------------------------
// TDM: 1-D tile copy global->LDS via Tensor Data Mover (D# per ISA ch.8)
// ---------------------------------------------------------------------------
__device__ __forceinline__ uint32_t lds_off32(const void* p) {
  // generic pointer to LDS: addr[31:0] is the LDS byte offset (aperture rule)
  return (uint32_t)(uintptr_t)p;
}

__device__ __forceinline__ void tdm_load_1d(uint32_t lds_addr, const void* gptr,
                                            uint32_t n_f32) {
  uint64_t ga = (uint64_t)(uintptr_t)gptr;
  v4u g0;
  g0[0] = 1u;                                            // count=1, user descriptor
  g0[1] = lds_addr;                                      // lds_addr [63:32]
  g0[2] = (uint32_t)(ga & 0xffffffffu);                  // global_addr[31:0]
  g0[3] = (uint32_t)((ga >> 32) & 0x01ffffffu) | (2u << 30); // global_addr[56:32] | type=2
  v8i g1;
  g1[0] = (int)(2u << 16);                               // data_size=4B, no flags
  g1[1] = (int)((n_f32 & 0xffffu) << 16);                // tensor_dim0[15:0]
  g1[2] = (int)((n_f32 >> 16) | (1u << 16));             // tensor_dim0[31:16], tensor_dim1=1
  g1[3] = (int)((n_f32 & 0xffffu) << 16);                // tile_dim0
  g1[4] = 1;                                             // tile_dim1=1, tile_dim2=0
  g1[5] = (int)n_f32;                                    // tensor_dim0_stride[31:0]
  g1[6] = 0;                                             // stride hi / dim1_stride lo
  g1[7] = 0;
  v4i z4 = {0, 0, 0, 0};                                 // groups 2/3 unused (<=2D tensor)
  v8i z8 = {0, 0, 0, 0, 0, 0, 0, 0};                     // extended group (6-arg toolchain)
  __builtin_amdgcn_tensor_load_to_lds(g0, g1, z4, z4, z8, 0);
}

// ---------------------------------------------------------------------------
// Kernel 0: zero the workspace accumulators (graph-capture safe)
// ---------------------------------------------------------------------------
__global__ void ccl_zero_ws(float* ws, int n) {
  int i = blockIdx.x * blockDim.x + threadIdx.x;
  if (i < n) ws[i] = 0.0f;
}

// ---------------------------------------------------------------------------
// Kernel 1: streaming pass.
//   - TDM double-buffered load of 32x128 f32 logit tiles into LDS
//   - per-wave softmax (32 lanes x 4 cols), write f16 probsT (transposed)
//   - per-segment n / Q partials via LDS atomics
//   - segmented sum s = OneHot^T x P as v_wmma_f32_16x16x32_f16
//     (each of 8 waves owns 16 segment rows x 128 prob cols = 8 acc tiles)
// ---------------------------------------------------------------------------
__global__ void __launch_bounds__(256)
ccl_main(const float* __restrict__ logits,
         const int* __restrict__ segids,
         const unsigned char* __restrict__ valid,
         float* __restrict__ ws_s,   // [128*128]
         float* __restrict__ ws_n,   // [128]
         float* __restrict__ ws_q,   // [128]
         int numBlk) {
  __shared__ __align__(16) float    lds_logits[2][TOK_BLK * NCOL];
  __shared__ __align__(16) _Float16 probsT[NCOL * PT_STR];
  __shared__ int   seg_lds[TOK_BLK];
  __shared__ float n_lds[NCOL];
  __shared__ float q_lds[NCOL];

  const int tid  = threadIdx.x;
  const int lane = tid & 31;
  const int wave = tid >> 5;

  if (tid < NCOL) { n_lds[tid] = 0.0f; q_lds[tid] = 0.0f; }

  v8f acc[8] = {};   // 8 output tiles: rows [wave*16, wave*16+16) x cols [j*16, j*16+16)

  // prime the TDM pipeline (wave 0 owns the DMA + TENSORcnt)
  const int firstBlk = blockIdx.x;
  if (wave == 0 && firstBlk < numBlk) {
    tdm_load_1d(lds_off32(&lds_logits[0][0]),
                logits + (size_t)firstBlk * TOK_BLK * NCOL, TOK_BLK * NCOL);
  }

  int it = 0;
  for (int blk = firstBlk; blk < numBlk; blk += gridDim.x, ++it) {
    const int cur = it & 1;
    const int nxt = blk + gridDim.x;
    if (wave == 0) {
      if (nxt < numBlk) {
        tdm_load_1d(lds_off32(&lds_logits[cur ^ 1][0]),
                    logits + (size_t)nxt * TOK_BLK * NCOL, TOK_BLK * NCOL);
        __builtin_amdgcn_s_wait_tensorcnt(1);   // current tile complete (in-order TDM)
      } else {
        __builtin_amdgcn_s_wait_tensorcnt(0);
      }
    }
    __syncthreads();   // current logits tile visible to all waves

    // ---- softmax: each wave handles 4 tokens, 32 lanes x 4 columns ----
    const float* lbuf = &lds_logits[cur][0];
    #pragma unroll
    for (int s4 = 0; s4 < 4; ++s4) {
      const int tt = wave * 4 + s4;
      v4f x = *(const v4f*)(lbuf + tt * NCOL + lane * 4);
      float m = fmaxf(fmaxf(x[0], x[1]), fmaxf(x[2], x[3]));
      #pragma unroll
      for (int d = 16; d >= 1; d >>= 1) m = fmaxf(m, __shfl_xor(m, d, 32));
      float e0 = __expf(x[0] - m), e1 = __expf(x[1] - m);
      float e2 = __expf(x[2] - m), e3 = __expf(x[3] - m);
      float ssum = e0 + e1 + e2 + e3;
      #pragma unroll
      for (int d = 16; d >= 1; d >>= 1) ssum += __shfl_xor(ssum, d, 32);
      const float inv = 1.0f / ssum;
      const float p0 = e0 * inv, p1 = e1 * inv, p2 = e2 * inv, p3 = e3 * inv;
      const int col = lane * 4;
      probsT[(col + 0) * PT_STR + tt] = (_Float16)p0;
      probsT[(col + 1) * PT_STR + tt] = (_Float16)p1;
      probsT[(col + 2) * PT_STR + tt] = (_Float16)p2;
      probsT[(col + 3) * PT_STR + tt] = (_Float16)p3;
      float q = p0 * p0 + p1 * p1 + p2 * p2 + p3 * p3;
      #pragma unroll
      for (int d = 16; d >= 1; d >>= 1) q += __shfl_xor(q, d, 32);
      if (lane == 0) {
        const size_t gtok = (size_t)blk * TOK_BLK + tt;
        const int sg = segids[gtok];
        const bool v = valid[gtok] != 0;
        seg_lds[tt] = v ? sg : -1;
        if (v) {
          atomicAdd(&n_lds[sg], 1.0f);
          atomicAdd(&q_lds[sg], q);
        }
      }
    }
    __syncthreads();   // probsT + seg_lds ready

    // ---- A fragment: one-hot of segment ids for this wave's 16 rows ----
    // 16-bit A 16x32 layout: lane<16 -> K = {0..7,16..23}; lane>=16 -> K = {8..15,24..31}
    const int rowM  = wave * 16 + (lane & 15);
    const int kHalf = (lane & 16) ? 8 : 0;
    v16h afrag;
    #pragma unroll
    for (int e = 0; e < 16; ++e) {
      const int k = kHalf + e + ((e >= 8) ? 8 : 0);
      afrag[e] = (_Float16)((seg_lds[k] == rowM) ? 1.0f : 0.0f);
    }

    // ---- 8 WMMAs: s[rows, j*16 + n] += A x P_tile ----
    // 16-bit B 32x16 layout: lane<16 holds K={0..7,16..23} of col N=lane; lane>=16 holds K={8..15,24..31}
    const int kOffB = (lane & 16) ? 8 : 0;
    #pragma unroll
    for (int j = 0; j < 8; ++j) {
      const int col = j * 16 + (lane & 15);
      const v8h* bp = (const v8h*)(probsT + col * PT_STR + kOffB);
      const v8h b0 = bp[0];
      const v8h b1 = bp[2];
      v16h bfrag;
      #pragma unroll
      for (int e = 0; e < 8; ++e) { bfrag[e] = b0[e]; bfrag[8 + e] = b1[e]; }
      acc[j] = __builtin_amdgcn_wmma_f32_16x16x32_f16(
          /*neg_a=*/false, afrag, /*neg_b=*/false, bfrag,
          /*c_mod=*/(short)0, acc[j], /*reuse_a=*/false, /*reuse_b=*/false);
    }
    // next iteration's first barrier separates these probsT reads from rewrites
  }

  // ---- flush per-WG partial s to global (C/D layout: VGPR r -> M = r (+8 for hi lanes)) ----
  const int rBase = wave * 16 + ((lane & 16) ? 8 : 0);
  const int colL  = lane & 15;
  #pragma unroll
  for (int j = 0; j < 8; ++j) {
    #pragma unroll
    for (int r = 0; r < 8; ++r) {
      atomicAdd(&ws_s[(rBase + r) * NCOL + j * 16 + colL], acc[j][r]);
    }
  }

  __syncthreads();
  if (tid < NCOL) {
    atomicAdd(&ws_n[tid], n_lds[tid]);
    atomicAdd(&ws_q[tid], q_lds[tid]);
  }
}

// ---------------------------------------------------------------------------
// Kernel 2: finalize.  var_c = (Q_c - |s_c|^2 / n_c) / (n_c * C); mean over n_c > 1
// ---------------------------------------------------------------------------
__global__ void ccl_finalize(const float* __restrict__ ws_s,
                             const float* __restrict__ ws_n,
                             const float* __restrict__ ws_q,
                             float* __restrict__ out) {
  __shared__ float var_s[NCOL];
  __shared__ float cnt_s[NCOL];
  const int c = threadIdx.x;
  const float n = ws_n[c];
  const float q = ws_q[c];
  float ss = 0.0f;
  #pragma unroll 4
  for (int j = 0; j < NCOL; ++j) {
    const float v = ws_s[c * NCOL + j];
    ss += v * v;
  }
  const float nsafe = fmaxf(n, 1.0f);
  float ssd = q - ss / nsafe;      // exact for n>=1; zero otherwise
  ssd = fmaxf(ssd, 0.0f);          // numerical guard
  const float var = ssd / (nsafe * (float)NCOL);
  const bool multi = n > 1.0f;
  var_s[c] = multi ? var : 0.0f;
  cnt_s[c] = multi ? 1.0f : 0.0f;
  __syncthreads();
  if (c == 0) {
    float tv = 0.0f, tc = 0.0f;
    for (int j = 0; j < NCOL; ++j) { tv += var_s[j]; tc += cnt_s[j]; }
    out[0] = (tc > 0.0f) ? (tv / tc) : 0.0f;
  }
}

// ---------------------------------------------------------------------------
extern "C" void kernel_launch(void* const* d_in, const int* in_sizes, int n_in,
                              void* d_out, int out_size, void* d_ws, size_t ws_size,
                              hipStream_t stream) {
  const float*         logits = (const float*)d_in[0];
  const int*           segp   = (const int*)d_in[1];
  const unsigned char* validp = (const unsigned char*)d_in[2];  // jnp bool -> 1 byte

  const int N      = in_sizes[1];          // B*T tokens
  const int numBlk = N / TOK_BLK;

  float* ws_s = (float*)d_ws;              // 128*128
  float* ws_n = ws_s + NCOL * NCOL;        // 128
  float* ws_q = ws_n + NCOL;               // 128

  const int zn = NCOL * NCOL + 2 * NCOL;
  ccl_zero_ws<<<(zn + 255) / 256, 256, 0, stream>>>(ws_s, zn);

  ccl_main<<<128, 256, 0, stream>>>(logits, segp, validp, ws_s, ws_n, ws_q, numBlk);

  ccl_finalize<<<1, NCOL, 0, stream>>>(ws_s, ws_n, ws_q, (float*)d_out);
}
